// HybridAttention_25443386261593
// MI455X (gfx1250) — compile-verified, matrix operand order checked by reading, not run on hardware
//
#include <hip/hip_runtime.h>
#include <hip/hip_bf16.h>

#define B_ 4
#define T_ 4096
#define C_ 512
#define WIN 8

typedef __attribute__((ext_vector_type(16))) __bf16 v16bf;
typedef __attribute__((ext_vector_type(8)))  float  v8f;

union Frag { v16bf v; unsigned short h[16]; uint4 q[2]; };

// fp32 -> bf16 round-to-nearest-even
__device__ __forceinline__ unsigned short f2bf(float f) {
  unsigned int u = __float_as_uint(f);
  u += 0x7fffu + ((u >> 16) & 1u);
  return (unsigned short)(u >> 16);
}

// load 16 contiguous bf16 (32 bytes) as a WMMA fragment register set
__device__ __forceinline__ v16bf load16(const unsigned short* p) {
  Frag f;
  f.q[0] = *(const uint4*)(p);
  f.q[1] = *(const uint4*)(p + 8);
  return f.v;
}

__device__ __forceinline__ v8f wmma_bf16(v16bf a, v16bf b, v8f c) {
  return __builtin_amdgcn_wmma_f32_16x16x32_bf16(false, a, false, b, (short)0, c,
                                                 false, false);
}

// ---------------- fp32 -> bf16 conversion (vectorized) ---
__global__ void k_f2bf(const float4* __restrict__ in, ushort4* __restrict__ out, int n4) {
  int i = blockIdx.x * blockDim.x + threadIdx.x;
  int stride = gridDim.x * blockDim.x;
  for (; i < n4; i += stride) {
    float4 f = in[i];
    ushort4 o;
    o.x = f2bf(f.x); o.y = f2bf(f.y); o.z = f2bf(f.z); o.w = f2bf(f.w);
    out[i] = o;
  }
}

// ---------------- fused QKV projection -------------------
// grid: (B*T/16, C/256), block: 128 (4 waves). Each wave: 16 rows x 64 cols
// of q,k,v.  Software-pipelined at (kc, weight) granularity: fragments for
// group g+1 load while group g's 4 WMMAs execute.
__global__ void __launch_bounds__(128) k_qkv(
    const unsigned short* __restrict__ xb,
    const unsigned short* __restrict__ wqb,
    const unsigned short* __restrict__ wkb,
    const unsigned short* __restrict__ wvb,
    const float* __restrict__ bq, const float* __restrict__ bk, const float* __restrict__ bv,
    unsigned short* __restrict__ Qb, unsigned short* __restrict__ Kb,
    unsigned short* __restrict__ Vtb)
{
  const int lane = threadIdx.x & 31;
  const int wave = threadIdx.x >> 5;
  const int n    = lane & 15;   // column-in-tile / row-in-A-tile
  const int hi   = lane >> 4;   // K-half selector

  const int rowblk = blockIdx.x;
  const int b  = rowblk / (T_ / 16);
  const int t0 = (rowblk % (T_ / 16)) * 16;
  const int n0 = blockIdx.y * 256 + wave * 64;

  const unsigned short* xrow = xb + (size_t)(b * T_ + t0) * C_;

  v8f acc[3][4] = {};   // [weight][nt]

  auto loadA = [&](int kc) {
    return load16(xrow + (size_t)n * C_ + kc * 32 + hi * 16);
  };
  auto loadG = [&](int g, v16bf fr[4]) {
    const int kc = g / 3, w = g % 3;
    const unsigned short* base = (w == 0) ? wqb : ((w == 1) ? wkb : wvb);
#pragma unroll
    for (int nt = 0; nt < 4; ++nt)
      fr[nt] = load16(base + (size_t)(n0 + nt * 16 + n) * C_ + kc * 32 + hi * 16);
  };

  const int NG = (C_ / 32) * 3;   // 48 groups
  v16bf curF[4], nxtF[4];
  v16bf aCur = loadA(0), aNxt;
  loadG(0, curF);

#pragma unroll
  for (int g = 0; g < NG; ++g) {
    const int w = g % 3;
    if (g + 1 < NG) {
      loadG(g + 1, nxtF);                     // issue next group's loads
      if ((g + 1) % 3 == 0) aNxt = loadA((g + 1) / 3);
    }
#pragma unroll
    for (int nt = 0; nt < 4; ++nt)
      acc[w][nt] = wmma_bf16(aCur, curF[nt], acc[w][nt]);
    if (g + 1 < NG) {
#pragma unroll
      for (int nt = 0; nt < 4; ++nt) curF[nt] = nxtF[nt];
      if ((g + 1) % 3 == 0) aCur = aNxt;
    }
  }

#pragma unroll
  for (int nt = 0; nt < 4; ++nt) {
    const int c = n0 + nt * 16 + n;
    const float biq = bq[c], bik = bk[c], biv = bv[c];
#pragma unroll
    for (int v = 0; v < 8; ++v) {
      const int m = v + 8 * hi;           // C/D layout: row = v + 8*(lane/16)
      const int t = t0 + m;
      const size_t ro = (size_t)(b * T_ + t) * C_ + c;
      Qb[ro] = f2bf(acc[0][nt][v] + biq);
      Kb[ro] = f2bf(acc[1][nt][v] + bik);
      Vtb[(size_t)(b * C_ + c) * T_ + t] = f2bf(acc[2][nt][v] + biv); // transposed
    }
  }
}

// ---------------- banded-causal flash attention ----------
// grid: B*T/16 blocks, 128 threads (4 waves). One 16-row query tile per block.
// Waves split the C=512 reduction of Q*K^T (128 each), reduce via LDS, then
// each wave owns 128 of the 512 output columns for P*V.  V fragments are
// issued before the softmax so their latency hides under VALU work.
__global__ void __launch_bounds__(128) k_attn(
    const unsigned short* __restrict__ Qb,
    const unsigned short* __restrict__ Kb,
    const unsigned short* __restrict__ Vtb,
    unsigned short* __restrict__ Ob)
{
  __shared__ __align__(16) unsigned short Qs[16 * C_];   // 16 KB
  __shared__ float Sred[4][16 * 32];                     // 8 KB partial scores
  __shared__ __align__(16) unsigned short Pbuf[16 * 32]; // 1 KB P in A-layout

  const int lane = threadIdx.x & 31;
  const int wave = threadIdx.x >> 5;
  const int n    = lane & 15;
  const int hi   = lane >> 4;

  const int blk = blockIdx.x;
  const int b   = blk / (T_ / 16);
  const int i0  = (blk % (T_ / 16)) * 16;

  { // stage Q tile (16 x 512 bf16) into LDS, 128 threads x 8 uint4 each
    const uint4* src = (const uint4*)(Qb + (size_t)(b * T_ + i0) * C_);
    uint4* dst = (uint4*)Qs;
    for (int i = threadIdx.x; i < 16 * C_ / 8; i += 128) dst[i] = src[i];
  }
  __syncthreads();

  v8f acc[8] = {};
  float mrow[8], lrow[8];
#pragma unroll
  for (int v = 0; v < 8; ++v) { mrow[v] = -1e30f; lrow[v] = 0.f; }

  const float scale = 0.044194173824159216f; // 1/sqrt(512)
  int jmax = i0 + 15 + WIN; if (jmax > T_ - 1) jmax = T_ - 1;
  const int nkb = jmax / 32 + 1;

  for (int kb = 0; kb < nkb; ++kb) {
    const int jb = kb * 32;

    // prefetch next key block's K rows (this wave's 128-col chunk)
    if (kb + 1 < nkb) {
      const unsigned short* pk =
          Kb + (size_t)(b * T_ + jb + 32 + lane) * C_ + wave * 128;
      __builtin_prefetch(pk, 0, 1);
      __builtin_prefetch(pk + 64, 0, 1);
    }

    // partial S = Q[:, c0:c0+128] * K[:, c0:c0+128]^T for 32 keys
    // (double-buffered over the 4 K-chunks)
    v8f s0p = {}, s1p = {};
    {
      auto ldA  = [&](int kc) {
        return load16(Qs + n * C_ + wave * 128 + kc * 32 + hi * 16);
      };
      auto ldB0 = [&](int kc) {
        return load16(Kb + (size_t)(b * T_ + jb + n) * C_ + wave * 128 + kc * 32 + hi * 16);
      };
      auto ldB1 = [&](int kc) {
        return load16(Kb + (size_t)(b * T_ + jb + 16 + n) * C_ + wave * 128 + kc * 32 + hi * 16);
      };
      v16bf aC = ldA(0), b0C = ldB0(0), b1C = ldB1(0);
      v16bf aN, b0N, b1N;
#pragma unroll
      for (int kc = 0; kc < 4; ++kc) {
        if (kc + 1 < 4) { aN = ldA(kc + 1); b0N = ldB0(kc + 1); b1N = ldB1(kc + 1); }
        s0p = wmma_bf16(aC, b0C, s0p);
        s1p = wmma_bf16(aC, b1C, s1p);
        if (kc + 1 < 4) { aC = aN; b0C = b0N; b1C = b1N; }
      }
    }

    // issue V fragment loads now: latency hides under reduce + softmax
    v16bf vf[8];
#pragma unroll
    for (int nt = 0; nt < 8; ++nt)
      vf[nt] = load16(Vtb + (size_t)(b * C_ + wave * 128 + nt * 16 + n) * T_ +
                      jb + hi * 16);

    { // deposit partial S into LDS
      float* dst = Sred[wave];
#pragma unroll
      for (int v = 0; v < 8; ++v) {
        const int m = v + 8 * hi;
        dst[m * 32 + n]      = s0p[v];
        dst[m * 32 + 16 + n] = s1p[v];
      }
    }
    __syncthreads();

    // cross-wave reduce + scale + band mask
    float s0[8], s1[8];
#pragma unroll
    for (int v = 0; v < 8; ++v) {
      const int m = v + 8 * hi;
      float a0 = 0.f, a1 = 0.f;
      for (int w = 0; w < 4; ++w) {
        a0 += Sred[w][m * 32 + n];
        a1 += Sred[w][m * 32 + 16 + n];
      }
      const int irow = i0 + m;
      s0[v] = (jb + n      <= irow + WIN) ? a0 * scale : -1e30f;
      s1[v] = (jb + 16 + n <= irow + WIN) ? a1 * scale : -1e30f;
    }

    // online softmax (row reductions stay inside each 16-lane half of wave32)
    float p0[8], p1[8];
#pragma unroll
    for (int v = 0; v < 8; ++v) {
      float mx = fmaxf(s0[v], s1[v]);
      mx = fmaxf(mx, __shfl_xor(mx, 1));
      mx = fmaxf(mx, __shfl_xor(mx, 2));
      mx = fmaxf(mx, __shfl_xor(mx, 4));
      mx = fmaxf(mx, __shfl_xor(mx, 8));
      const float mnew = fmaxf(mrow[v], mx);
      const float sc = __expf(mrow[v] - mnew);
      mrow[v] = mnew;
      p0[v] = __expf(s0[v] - mnew);
      p1[v] = __expf(s1[v] - mnew);
      float rs = p0[v] + p1[v];
      rs += __shfl_xor(rs, 1);
      rs += __shfl_xor(rs, 2);
      rs += __shfl_xor(rs, 4);
      rs += __shfl_xor(rs, 8);
      lrow[v] = lrow[v] * sc + rs;
#pragma unroll
      for (int nt = 0; nt < 8; ++nt) acc[nt][v] *= sc;
    }

    // wave 0 publishes P (bf16) in A-fragment [m][k] layout
    if (wave == 0) {
#pragma unroll
      for (int v = 0; v < 8; ++v) {
        const int m = v + 8 * hi;
        Pbuf[m * 32 + n]      = f2bf(p0[v]);
        Pbuf[m * 32 + 16 + n] = f2bf(p1[v]);
      }
    }
    __syncthreads();

    // O[:, wave*128 : wave*128+128] += P * V  (V fragments already in regs)
    v16bf pa = load16(Pbuf + n * 32 + hi * 16);
#pragma unroll
    for (int nt = 0; nt < 8; ++nt)
      acc[nt] = wmma_bf16(pa, vf[nt], acc[nt]);
  }

  // normalize + store O as bf16 for the final projection
#pragma unroll
  for (int nt = 0; nt < 8; ++nt) {
    const int c = wave * 128 + nt * 16 + n;
#pragma unroll
    for (int v = 0; v < 8; ++v) {
      const int m = v + 8 * hi;
      Ob[(size_t)(b * T_ + i0 + m) * C_ + c] = f2bf(acc[nt][v] / lrow[v]);
    }
  }
}

// ---------------- output projection ----------------------
__global__ void __launch_bounds__(128) k_oproj(
    const unsigned short* __restrict__ Ob,
    const unsigned short* __restrict__ wob,
    const float* __restrict__ bo,
    float* __restrict__ out)
{
  const int lane = threadIdx.x & 31;
  const int wave = threadIdx.x >> 5;
  const int n    = lane & 15;
  const int hi   = lane >> 4;

  const int rowblk = blockIdx.x;
  const int b  = rowblk / (T_ / 16);
  const int t0 = (rowblk % (T_ / 16)) * 16;
  const int n0 = blockIdx.y * 256 + wave * 64;

  const unsigned short* orow = Ob + (size_t)(b * T_ + t0) * C_;
  v8f acc[4] = {};

  auto loadA = [&](int kc) {
    return load16(orow + (size_t)n * C_ + kc * 32 + hi * 16);
  };
  auto loadW = [&](int kc, v16bf fr[4]) {
#pragma unroll
    for (int nt = 0; nt < 4; ++nt)
      fr[nt] = load16(wob + (size_t)(n0 + nt * 16 + n) * C_ + kc * 32 + hi * 16);
  };

  const int KC = C_ / 32;
  v16bf curF[4], nxtF[4];
  v16bf aCur = loadA(0), aNxt;
  loadW(0, curF);

#pragma unroll
  for (int kc = 0; kc < KC; ++kc) {
    if (kc + 1 < KC) { loadW(kc + 1, nxtF); aNxt = loadA(kc + 1); }
#pragma unroll
    for (int nt = 0; nt < 4; ++nt)
      acc[nt] = wmma_bf16(aCur, curF[nt], acc[nt]);
    if (kc + 1 < KC) {
#pragma unroll
      for (int nt = 0; nt < 4; ++nt) curF[nt] = nxtF[nt];
      aCur = aNxt;
    }
  }

#pragma unroll
  for (int nt = 0; nt < 4; ++nt) {
    const int c = n0 + nt * 16 + n;
    const float bias = bo[c];
#pragma unroll
    for (int v = 0; v < 8; ++v) {
      const int m = v + 8 * hi;
      out[(size_t)(b * T_ + t0 + m) * C_ + c] = acc[nt][v] + bias;
    }
  }
}

extern "C" void kernel_launch(void* const* d_in, const int* in_sizes, int n_in,
                              void* d_out, int out_size, void* d_ws, size_t ws_size,
                              hipStream_t stream) {
  const float* x  = (const float*)d_in[0];
  const float* wq = (const float*)d_in[1];
  const float* bq = (const float*)d_in[2];
  const float* wk = (const float*)d_in[3];
  const float* bk = (const float*)d_in[4];
  const float* wv = (const float*)d_in[5];
  const float* bv = (const float*)d_in[6];
  const float* wo = (const float*)d_in[7];
  const float* bo = (const float*)d_in[8];
  float* out = (float*)d_out;

  char* ws = (char*)d_ws;
  size_t off = 0;
  auto take = [&](size_t bytes) {
    void* p = ws + off;
    off += (bytes + 255) & ~(size_t)255;
    return p;
  };
  unsigned short* xb  = (unsigned short*)take((size_t)B_ * T_ * C_ * 2);
  unsigned short* wqb = (unsigned short*)take((size_t)C_ * C_ * 2);
  unsigned short* wkb = (unsigned short*)take((size_t)C_ * C_ * 2);
  unsigned short* wvb = (unsigned short*)take((size_t)C_ * C_ * 2);
  unsigned short* wob = (unsigned short*)take((size_t)C_ * C_ * 2);
  unsigned short* Qb  = (unsigned short*)take((size_t)B_ * T_ * C_ * 2);
  unsigned short* Kb  = (unsigned short*)take((size_t)B_ * T_ * C_ * 2);
  unsigned short* Vtb = (unsigned short*)take((size_t)B_ * T_ * C_ * 2);
  unsigned short* Ob  = (unsigned short*)take((size_t)B_ * T_ * C_ * 2);
  (void)ws_size; (void)in_sizes; (void)n_in; (void)out_size;

  k_f2bf<<<512, 256, 0, stream>>>((const float4*)x,  (ushort4*)xb,  B_ * T_ * C_ / 4);
  k_f2bf<<<128, 256, 0, stream>>>((const float4*)wq, (ushort4*)wqb, C_ * C_ / 4);
  k_f2bf<<<128, 256, 0, stream>>>((const float4*)wk, (ushort4*)wkb, C_ * C_ / 4);
  k_f2bf<<<128, 256, 0, stream>>>((const float4*)wv, (ushort4*)wvb, C_ * C_ / 4);
  k_f2bf<<<128, 256, 0, stream>>>((const float4*)wo, (ushort4*)wob, C_ * C_ / 4);

  dim3 gproj(B_ * T_ / 16, C_ / 256);
  k_qkv<<<gproj, 128, 0, stream>>>(xb, wqb, wkb, wvb, bq, bk, bv, Qb, Kb, Vtb);
  k_attn<<<B_ * T_ / 16, 128, 0, stream>>>(Qb, Kb, Vtb, Ob);
  k_oproj<<<gproj, 128, 0, stream>>>(Ob, wob, bo, out);
}
